// Encoder_70987219468959
// MI455X (gfx1250) — compile-verified, matrix operand order checked by reading, not run on hardware
//
#include <hip/hip_runtime.h>
#include <hip/hip_bf16.h>

typedef __attribute__((ext_vector_type(16))) __bf16 v16bf;
typedef __attribute__((ext_vector_type(8)))  float  v8f;

static constexpr int kV  = 32000;
static constexpr int kH  = 1024;
static constexpr int kS  = 4096;
static constexpr int kN  = 4 * kH;     // 4096 gate rows
static constexpr int kKT = kH / 32;    // 32 K-tiles of 32
static constexpr int kMT = kS / 16;    // 256 M-tiles
static constexpr int kNT = kN / 16;    // 256 N-tiles

// round-to-nearest-even f32 -> bf16 (bit pattern as ushort)
__device__ __forceinline__ unsigned short f32_to_bf16_rne(float f) {
  unsigned u = __float_as_uint(f);
  u += 0x7FFFu + ((u >> 16) & 1u);
  return (unsigned short)(u >> 16);
}

// K-position inside a 32-wide tile for 16-bit WMMA A/B fragment element e of `lane`
// (ISA 7.12.2: lane<16 holds K 0..7 / 16..23, lane>=16 holds K 8..15 / 24..31)
__device__ __forceinline__ int frag_k(int lane, int e) {
  return ((lane >> 4) << 3) + ((e >> 3) << 4) + (e & 7);
}

__device__ __forceinline__ v8f wmma_bf16(v16bf a, v16bf b, v8f c) {
  return __builtin_amdgcn_wmma_f32_16x16x32_bf16(
      /*neg_a=*/false, a, /*neg_b=*/false, b,
      /*c_mod=*/(short)0, c, /*reuse_a=*/false, /*reuse_b=*/false);
}

// ---------------------------------------------------------------------------
// Pack A: gather embedding rows via x[] and emit bf16 WMMA A-fragments.
// packedA element index = ((mt*kKT + kt)*32 + lane)*16 + e
// ---------------------------------------------------------------------------
__global__ __launch_bounds__(256) void pack_a_kernel(
    const int* __restrict__ x, const float* __restrict__ emb,
    unsigned short* __restrict__ pA) {
  unsigned idx = blockIdx.x * 256u + threadIdx.x;   // < 4,194,304
  int e    = idx & 15;
  int lane = (idx >> 4) & 31;
  int kt   = (idx >> 9) & 31;
  int mt   = idx >> 14;
  int m = mt * 16 + (lane & 15);
  int k = kt * 32 + frag_k(lane, e);
  float v = emb[(size_t)x[m] * kH + k];
  pA[idx] = f32_to_bf16_rne(v);
}

// ---------------------------------------------------------------------------
// Pack B: B[k][n] = W_ih[n][k] as bf16 WMMA B-fragments.
// packedB element index = ((kt*kNT + nt)*32 + lane)*16 + e
// ---------------------------------------------------------------------------
__global__ __launch_bounds__(256) void pack_b_kernel(
    const float* __restrict__ Wih, unsigned short* __restrict__ pB) {
  unsigned idx = blockIdx.x * 256u + threadIdx.x;   // < 4,194,304
  int e    = idx & 15;
  int lane = (idx >> 4) & 31;
  int nt   = (idx >> 9) & 255;
  int kt   = idx >> 17;
  int n = nt * 16 + (lane & 15);
  int k = kt * 32 + frag_k(lane, e);
  pB[idx] = f32_to_bf16_rne(Wih[(size_t)n * kH + k]);
}

// ---------------------------------------------------------------------------
// GEMM: xg[S,4H] = A(bf16) x B(bf16) + (b_ih + b_hh), f32 accumulate via WMMA.
// Block = 8 waves; wave w owns 2 M-tiles x 4 N-tiles (8 accumulators).
// Fragment loads are software-pipelined one K-step ahead so the 8 WMMAs of
// step kt overlap the L2 loads for step kt+1 (no per-WMMA s_wait_loadcnt 0).
// ---------------------------------------------------------------------------
__global__ __launch_bounds__(256) void gemm_xg_kernel(
    const unsigned short* __restrict__ pA,
    const unsigned short* __restrict__ pB,
    const float* __restrict__ b_ih, const float* __restrict__ b_hh,
    float* __restrict__ xg) {
  const int lane = threadIdx.x & 31;
  const int wave = threadIdx.x >> 5;
  const int mt0  = blockIdx.x * 16 + wave * 2;  // two consecutive M-tiles
  const int nt0  = blockIdx.y * 4;              // four consecutive N-tiles

  // A frag (mi, kt) at Af[mi*kKT*32 + kt*32]; B frag (j, kt) at Bf[kt*kNT*32 + j*32]
  const v16bf* Af = (const v16bf*)pA + (size_t)mt0 * (kKT * 32) + lane;
  const v16bf* Bf = (const v16bf*)pB + (size_t)nt0 * 32 + lane;

  v8f acc[2][4] = {};

  // prologue: load kt = 0 fragments
  v16bf a0 = Af[0];
  v16bf a1 = Af[kKT * 32];
  v16bf b0 = Bf[0];
  v16bf b1 = Bf[32];
  v16bf b2 = Bf[64];
  v16bf b3 = Bf[96];

  for (int kt = 0; kt < kKT - 1; ++kt) {
    // prefetch kt+1 into fresh registers (overlaps the WMMAs below)
    const v16bf na0 = Af[(kt + 1) * 32];
    const v16bf na1 = Af[kKT * 32 + (kt + 1) * 32];
    const v16bf* Bk = Bf + (size_t)(kt + 1) * (kNT * 32);
    const v16bf nb0 = Bk[0];
    const v16bf nb1 = Bk[32];
    const v16bf nb2 = Bk[64];
    const v16bf nb3 = Bk[96];

    acc[0][0] = wmma_bf16(a0, b0, acc[0][0]);
    acc[0][1] = wmma_bf16(a0, b1, acc[0][1]);
    acc[0][2] = wmma_bf16(a0, b2, acc[0][2]);
    acc[0][3] = wmma_bf16(a0, b3, acc[0][3]);
    acc[1][0] = wmma_bf16(a1, b0, acc[1][0]);
    acc[1][1] = wmma_bf16(a1, b1, acc[1][1]);
    acc[1][2] = wmma_bf16(a1, b2, acc[1][2]);
    acc[1][3] = wmma_bf16(a1, b3, acc[1][3]);

    a0 = na0; a1 = na1; b0 = nb0; b1 = nb1; b2 = nb2; b3 = nb3;
  }
  // epilogue: last K-step
  acc[0][0] = wmma_bf16(a0, b0, acc[0][0]);
  acc[0][1] = wmma_bf16(a0, b1, acc[0][1]);
  acc[0][2] = wmma_bf16(a0, b2, acc[0][2]);
  acc[0][3] = wmma_bf16(a0, b3, acc[0][3]);
  acc[1][0] = wmma_bf16(a1, b0, acc[1][0]);
  acc[1][1] = wmma_bf16(a1, b1, acc[1][1]);
  acc[1][2] = wmma_bf16(a1, b2, acc[1][2]);
  acc[1][3] = wmma_bf16(a1, b3, acc[1][3]);

  // store with fused bias. C/D layout: N = lane&15, M = v + 8*(lane>>4)
#pragma unroll
  for (int mi = 0; mi < 2; ++mi) {
    const int mrow0 = (mt0 + mi) * 16 + ((lane >> 4) << 3);
#pragma unroll
    for (int j = 0; j < 4; ++j) {
      const int n = (nt0 + j) * 16 + (lane & 15);
      const float bias = b_ih[n] + b_hh[n];
#pragma unroll
      for (int v = 0; v < 8; ++v)
        xg[(size_t)(mrow0 + v) * kN + n] = acc[mi][j][v] + bias;
    }
  }
}

// ---------------------------------------------------------------------------
// Persistent LSTM scan. 64 blocks x 256 threads, 16 lanes per hidden unit j.
// h double-buffered in global; c lives in a register of lane 0 of each group.
// Steps separated by a monotonic atomic grid barrier.
// ---------------------------------------------------------------------------
__global__ __launch_bounds__(256) void lstm_scan_kernel(
    const float* __restrict__ Whh, const float* __restrict__ xg,
    float* __restrict__ hbuf,        // [2][kH], zeroed before launch
    unsigned* __restrict__ bar,      // zeroed before launch
    float* __restrict__ out) {       // [kH + kS*kH]
  __shared__ float hs[kH];
  const int tid = threadIdx.x;
  const int grp = tid >> 4;          // 0..15
  const int l16 = tid & 15;
  const int j   = blockIdx.x * 16 + grp;   // hidden index, 0..1023
  const unsigned nblk = gridDim.x;

  const float4* w0 = (const float4*)(Whh + (size_t)(0 * kH + j) * kH);
  const float4* w1 = (const float4*)(Whh + (size_t)(1 * kH + j) * kH);
  const float4* w2 = (const float4*)(Whh + (size_t)(2 * kH + j) * kH);
  const float4* w3 = (const float4*)(Whh + (size_t)(3 * kH + j) * kH);

  float c = 0.0f;

  for (int t = 0; t < kS; ++t) {
    if (t) {  // wait for every block to finish step t-1
      __syncthreads();
      if (tid == 0) {
        __hip_atomic_fetch_add(bar, 1u, __ATOMIC_RELEASE, __HIP_MEMORY_SCOPE_AGENT);
        const unsigned want = (unsigned)t * nblk;
        while (__hip_atomic_load(bar, __ATOMIC_ACQUIRE, __HIP_MEMORY_SCOPE_AGENT) < want)
          __builtin_amdgcn_s_sleep(2);
      }
      __syncthreads();
    }
    // stage h(t) into LDS
    const float* hsrc = hbuf + (size_t)(t & 1) * kH;
    for (int i = tid; i < kH; i += 256) hs[i] = hsrc[i];
    __syncthreads();

    float a0 = 0.f, a1 = 0.f, a2 = 0.f, a3 = 0.f;
#pragma unroll 4
    for (int kk = 0; kk < 16; ++kk) {
      const int k4 = kk * 16 + l16;             // lanes -> consecutive float4s
      const float4 hv = ((const float4*)hs)[k4];
      const float4 v0 = w0[k4];
      const float4 v1 = w1[k4];
      const float4 v2 = w2[k4];
      const float4 v3 = w3[k4];
      a0 += v0.x * hv.x + v0.y * hv.y + v0.z * hv.z + v0.w * hv.w;
      a1 += v1.x * hv.x + v1.y * hv.y + v1.z * hv.z + v1.w * hv.w;
      a2 += v2.x * hv.x + v2.y * hv.y + v2.z * hv.z + v2.w * hv.w;
      a3 += v3.x * hv.x + v3.y * hv.y + v3.z * hv.z + v3.w * hv.w;
    }
    // reduce across the 16-lane group
    for (int m = 8; m; m >>= 1) {
      a0 += __shfl_xor(a0, m, 16);
      a1 += __shfl_xor(a1, m, 16);
      a2 += __shfl_xor(a2, m, 16);
      a3 += __shfl_xor(a3, m, 16);
    }
    if (l16 == 0) {
      const float* xr = xg + (size_t)t * kN;
      const float gi = a0 + xr[j];
      const float gf = a1 + xr[kH + j];
      const float gg = a2 + xr[2 * kH + j];
      const float go = a3 + xr[3 * kH + j];
      const float i_ = 1.0f / (1.0f + __expf(-gi));
      const float f_ = 1.0f / (1.0f + __expf(-gf));
      const float g_ = tanhf(gg);
      const float o_ = 1.0f / (1.0f + __expf(-go));
      c = f_ * c + i_ * g_;
      const float hn = o_ * tanhf(c);
      hbuf[(size_t)((t + 1) & 1) * kH + j] = hn;   // write buffer != read buffer
      out[kH + (size_t)t * kH + j] = hn;           // outs[t]
      if (t == kS - 1) out[j] = hn;                // final h
      __threadfence();
    }
    __syncthreads();
  }
}

// ---------------------------------------------------------------------------
extern "C" void kernel_launch(void* const* d_in, const int* in_sizes, int n_in,
                              void* d_out, int out_size, void* d_ws, size_t ws_size,
                              hipStream_t stream) {
  (void)in_sizes; (void)n_in; (void)out_size; (void)ws_size;
  const int*   x   = (const int*)d_in[0];
  const float* emb = (const float*)d_in[1];
  const float* Wih = (const float*)d_in[2];
  const float* Whh = (const float*)d_in[3];
  const float* bih = (const float*)d_in[4];
  const float* bhh = (const float*)d_in[5];
  float* out = (float*)d_out;

  char* ws = (char*)d_ws;
  unsigned*       bar  = (unsigned*)(ws + 0);        // 1 counter
  float*          hbuf = (float*)(ws + 256);         // 2*kH floats = 8 KB
  float*          xg   = (float*)(ws + 16384);       // kS*kN f32 = 64 MB
  unsigned short* pA   = (unsigned short*)(ws + 16384 + (size_t)kS * kN * 4);  // 8 MB
  unsigned short* pB   = pA + (size_t)kS * kH;                                  // 8 MB

  hipMemsetAsync(d_ws, 0, 16384, stream);  // barrier counter + h double-buffer = 0

  pack_a_kernel<<<(kS * kH) / 256, 256, 0, stream>>>(x, emb, pA);
  pack_b_kernel<<<(kH * kN) / 256, 256, 0, stream>>>(Wih, pB);
  gemm_xg_kernel<<<dim3(kMT / 16, kNT / 4), 256, 0, stream>>>(pA, pB, bih, bhh, xg);
  lstm_scan_kernel<<<64, 256, 0, stream>>>(Whh, xg, hbuf, bar, out);
}
